// MLAAttention_50843822850162
// MI455X (gfx1250) — matrix operand order repaired and checked by reading, not compiled, not run on hardware
//
#include <hip/hip_runtime.h>

// ---------------------------------------------------------------------------
// MI455X (gfx1250) GQA attention, bf16 WMMA pipeline.
// Wave32; WMMA 16x16x32 bf16 -> f32 accumulate throughout.
// ---------------------------------------------------------------------------

typedef __attribute__((ext_vector_type(16))) __bf16 v16bf;
typedef __attribute__((ext_vector_type(8)))  __bf16 v8bf;
typedef __attribute__((ext_vector_type(4)))  __bf16 v4bf;
typedef __attribute__((ext_vector_type(8)))  float  v8f;
typedef __attribute__((ext_vector_type(4)))  float  v4f;

#define B_    2
#define S_    2048
#define H_    1024
#define G_    4
#define HPG_  4
#define DG_   256
#define HD_   64
#define K_    1024           // contraction dim for all projection GEMMs
#define MTOT_ 4096           // B_*S_

// SCALE * log2(e) folded into Q so softmax is pure exp2 (v_exp_f32)
#define QSCALE 0.18033688011112042f   // (1/sqrt(64)) * 1.44269504088896340736

__device__ inline v8f wmma_bf16(v16bf a, v16bf b, v8f c) {
  return __builtin_amdgcn_wmma_f32_16x16x32_bf16(
      /*neg_a=*/false, a, /*neg_b=*/false, b,
      /*c_mod=*/(short)0, c, /*reuse_a=*/false, /*reuse_b=*/false);
}

__device__ inline v16bf pack16(v8bf lo, v8bf hi8) {
  v16bf r;
#pragma unroll
  for (int i = 0; i < 8; ++i) { r[i] = lo[i]; r[i + 8] = hi8[i]; }
  return r;
}

// A-fragment (16x32, 16-bit): lane holds row (lane&15); K-elements
// {0..15} + 16*(lane>>4)  -> two 16B contiguous loads from a bf16 row.
__device__ inline v16bf load_a_frag(const __bf16* rowptr, int hi) {
  v8bf lo  = *(const v8bf*)(rowptr + 16 * hi);
  v8bf hi8 = *(const v8bf*)(rowptr + 16 * hi + 8);
  return pack16(lo, hi8);
}

// ---------------------------------------------------------------------------
// fp32 -> bf16 conversion (vectorized 4/lane)
// ---------------------------------------------------------------------------
__global__ __launch_bounds__(256) void k_convert(const float* __restrict__ in,
                                                 __bf16* __restrict__ out, int n4) {
  int i = blockIdx.x * 256 + threadIdx.x;
  if (i < n4) {
    v4f v = *(const v4f*)(in + (size_t)i * 4);
    v4bf o;
#pragma unroll
    for (int j = 0; j < 4; ++j) o[j] = (__bf16)v[j];
    *(v4bf*)(out + (size_t)i * 4) = o;
  }
}

// ---------------------------------------------------------------------------
// GEMM: C[M,N] = A[M,K] * Bsrc[N,K]^T + bias[N]
// 256 threads = 8 waves; block tile 128x128, wave tile 32x64, K-step 32.
// MODE 0: out fp32 [MTOT,H]          (final projection + bo)
// MODE 1: out bf16 [B,G,S,DG], *QSCALE  (Q)
// MODE 2: out bf16 [B,G,S,DG]           (K)
// MODE 3: out bf16 [B,G,DG,S] transposed (V^T, so P*V B-frags are contiguous)
// ---------------------------------------------------------------------------
template <int MODE>
__global__ __launch_bounds__(256) void k_gemm(const __bf16* __restrict__ A,
                                              const __bf16* __restrict__ Bsrc,
                                              const float* __restrict__ bias,
                                              void* __restrict__ outp) {
  const int tid  = threadIdx.x;
  const int wave = tid >> 5;
  const int lane = tid & 31;
  const int col  = lane & 15;
  const int hi   = lane >> 4;
  const int wm   = wave >> 1;       // 0..3  (M)
  const int wn   = wave & 1;        // 0..1  (N)
  const int mBase = blockIdx.y * 128 + wm * 32;
  const int nBase = blockIdx.x * 128 + wn * 64;

  v8f acc[2][4] = {};

  for (int k0 = 0; k0 < K_; k0 += 32) {
    v16bf a[2];
#pragma unroll
    for (int mf = 0; mf < 2; ++mf) {
      const __bf16* p = A + (size_t)(mBase + mf * 16 + col) * K_ + k0;
      a[mf] = load_a_frag(p, hi);
    }
    if (k0 + 32 < K_) {   // speculative prefetch of next A K-slice
      __builtin_prefetch(A + (size_t)(mBase + col) * K_ + k0 + 32, 0, 0);
    }
#pragma unroll
    for (int nf = 0; nf < 4; ++nf) {
      // B-fragment: lane holds col (lane&15); K = e + 16*(lane>>4) -> 32B load
      const __bf16* bp = Bsrc + (size_t)(nBase + nf * 16 + col) * K_ + k0 + 16 * hi;
      v16bf bfrag = *(const v16bf*)bp;
#pragma unroll
      for (int mf = 0; mf < 2; ++mf)
        acc[mf][nf] = wmma_bf16(a[mf], bfrag, acc[mf][nf]);
    }
  }

  // ---- epilogue ----
#pragma unroll
  for (int nf = 0; nf < 4; ++nf) {
    const int n  = nBase + nf * 16 + col;
    const float bn = bias[n];
#pragma unroll
    for (int mf = 0; mf < 2; ++mf) {
      const int mTile = mBase + mf * 16;
      if constexpr (MODE == 0) {
        float* out = (float*)outp;
#pragma unroll
        for (int r = 0; r < 8; ++r) {
          int m = mTile + r + 8 * hi;
          out[(size_t)m * H_ + n] = acc[mf][nf][r] + bn;
        }
      } else if constexpr (MODE == 1 || MODE == 2) {
        __bf16* out = (__bf16*)outp;
        int g = n >> 8, d = n & 255;
#pragma unroll
        for (int r = 0; r < 8; ++r) {
          int m = mTile + r + 8 * hi;
          int b = m >> 11, s = m & (S_ - 1);
          float v = acc[mf][nf][r] + bn;
          if constexpr (MODE == 1) v *= QSCALE;
          out[(((size_t)(b * G_ + g) * S_ + s) * DG_) + d] = (__bf16)v;
        }
      } else {  // MODE 3: V^T [B,G,DG,S]; 8 consecutive s -> one 16B store
        __bf16* out = (__bf16*)outp;
        int g = n >> 8, d = n & 255;
        int b = mTile >> 11;
        int s0 = (mTile & (S_ - 1)) + 8 * hi;
        v8bf pk;
#pragma unroll
        for (int r = 0; r < 8; ++r) pk[r] = (__bf16)(acc[mf][nf][r] + bn);
        *(v8bf*)(out + ((size_t)(b * G_ + g) * DG_ + d) * S_ + s0) = pk;
      }
    }
  }
}

// ---------------------------------------------------------------------------
// Flash attention. grid = (S/64, B*G*HPG); 128 threads = 4 waves.
// Each wave owns one 16-row Q tile, streams KV in tiles of 32.
// Q pre-scaled by SCALE*log2e -> softmax with raw exp2.
// P tile transposed C-layout -> A-layout through bank-skewed LDS (stride 80B).
// ---------------------------------------------------------------------------
#define PSTRIDE 40   // bf16 elements per P row in LDS (80 B, 16B-aligned, skewed)

__global__ __launch_bounds__(128) void k_attn(const __bf16* __restrict__ Q,
                                              const __bf16* __restrict__ K,
                                              const __bf16* __restrict__ Vt,
                                              __bf16* __restrict__ ctx) {
  __shared__ __bf16 ldsP[4][16 * PSTRIDE];

  const int tid  = threadIdx.x;
  const int wave = tid >> 5;
  const int lane = tid & 31;
  const int col  = lane & 15;
  const int hi   = lane >> 4;

  const int head = blockIdx.y;
  const int hh = head & 3, g = (head >> 2) & 3, b = head >> 4;
  const int q0 = blockIdx.x * 64 + wave * 16;

  const __bf16* Qh = Q  + (size_t)(b * G_ + g) * S_ * DG_ + hh * HD_;
  const __bf16* Kh = K  + (size_t)(b * G_ + g) * S_ * DG_ + hh * HD_;
  const __bf16* Vh = Vt + ((size_t)(b * G_ + g) * DG_ + hh * HD_) * S_;

  // Q fragments for this wave's 16 rows (contraction hd=64 -> 2 frags)
  v16bf qa[2];
  {
    const __bf16* qrow = Qh + (size_t)(q0 + col) * DG_;
#pragma unroll
    for (int kf = 0; kf < 2; ++kf) qa[kf] = load_a_frag(qrow + kf * 32, hi);
  }

  v8f acc[4] = {};
  float mrow[8], lrow[8];
#pragma unroll
  for (int r = 0; r < 8; ++r) { mrow[r] = -1e30f; lrow[r] = 0.0f; }

  __bf16* pw = &ldsP[wave][0];

  for (int kv0 = 0; kv0 < S_; kv0 += 32) {
    // ---- scores: S[16q x 32kv] = Q(16x64) . K_tile(64x32 as B-frags)
    v8f sc[2] = {};
#pragma unroll
    for (int nt = 0; nt < 2; ++nt) {
      const __bf16* krow = Kh + (size_t)(kv0 + nt * 16 + col) * DG_;
#pragma unroll
      for (int kf = 0; kf < 2; ++kf) {
        v16bf bfrag = *(const v16bf*)(krow + kf * 32 + 16 * hi);
        sc[nt] = wmma_bf16(qa[kf], bfrag, sc[nt]);
      }
    }

    // ---- online softmax: row stats across the 16 lanes of each half-wave
    float rmax[8], rs[8];
#pragma unroll
    for (int r = 0; r < 8; ++r) rmax[r] = fmaxf(sc[0][r], sc[1][r]);
#pragma unroll
    for (int m = 1; m < 16; m <<= 1)
#pragma unroll
      for (int r = 0; r < 8; ++r) rmax[r] = fmaxf(rmax[r], __shfl_xor(rmax[r], m));

#pragma unroll
    for (int r = 0; r < 8; ++r) {
      float mnew  = fmaxf(mrow[r], rmax[r]);
      float alpha = __builtin_amdgcn_exp2f(mrow[r] - mnew);
      mrow[r] = mnew;
      float p0 = __builtin_amdgcn_exp2f(sc[0][r] - mnew);
      float p1 = __builtin_amdgcn_exp2f(sc[1][r] - mnew);
      sc[0][r] = p0; sc[1][r] = p1;
      rs[r] = p0 + p1;
      lrow[r] *= alpha;
#pragma unroll
      for (int f = 0; f < 4; ++f) acc[f][r] *= alpha;
    }
#pragma unroll
    for (int m = 1; m < 16; m <<= 1)
#pragma unroll
      for (int r = 0; r < 8; ++r) rs[r] += __shfl_xor(rs[r], m);
#pragma unroll
    for (int r = 0; r < 8; ++r) lrow[r] += rs[r];

    // ---- P: C-layout -> A-layout via wave-private LDS (in-order DS pipe)
#pragma unroll
    for (int r = 0; r < 8; ++r) {
      int m = r + 8 * hi;
      pw[m * PSTRIDE + col]      = (__bf16)sc[0][r];
      pw[m * PSTRIDE + 16 + col] = (__bf16)sc[1][r];
    }
    v16bf pa = load_a_frag(pw + col * PSTRIDE, hi);

    // ---- ctx += P(16x32) . V_tile(32x64); V^T rows are contiguous in kv
#pragma unroll
    for (int f = 0; f < 4; ++f) {
      const __bf16* vp = Vh + (size_t)(f * 16 + col) * S_ + kv0 + 16 * hi;
      v16bf vfrag = *(const v16bf*)vp;
      acc[f] = wmma_bf16(pa, vfrag, acc[f]);
    }
  }

  // ---- normalize and store ctx[b, s, (g*HPG+hh)*64 + hd] as bf16
  const size_t cbase = ((size_t)b * S_) * H_ + (size_t)(g * HPG_ + hh) * HD_;
#pragma unroll
  for (int r = 0; r < 8; ++r) {
    int s = q0 + r + 8 * hi;
    float inv = __builtin_amdgcn_rcpf(lrow[r]);
#pragma unroll
    for (int f = 0; f < 4; ++f)
      ctx[cbase + (size_t)s * H_ + f * 16 + col] = (__bf16)(acc[f][r] * inv);
  }
}

// ---------------------------------------------------------------------------
// Workspace layout (bytes), total ~48 MiB
// ---------------------------------------------------------------------------
#define OFF_XB   ((size_t)0)                       // x bf16      8 MiB
#define OFF_WQB  (OFF_XB  + (size_t)8388608)       // Wq bf16     2 MiB
#define OFF_WKB  (OFF_WQB + (size_t)2097152)
#define OFF_WVB  (OFF_WKB + (size_t)2097152)
#define OFF_WOB  (OFF_WVB + (size_t)2097152)
#define OFF_QB   (OFF_WOB + (size_t)2097152)       // Q bf16      8 MiB
#define OFF_KB   (OFF_QB  + (size_t)8388608)
#define OFF_VTB  (OFF_KB  + (size_t)8388608)
#define OFF_CTX  (OFF_VTB + (size_t)8388608)       // ctx bf16    8 MiB

extern "C" void kernel_launch(void* const* d_in, const int* in_sizes, int n_in,
                              void* d_out, int out_size, void* d_ws, size_t ws_size,
                              hipStream_t stream) {
  const float* x  = (const float*)d_in[0];
  const float* Wq = (const float*)d_in[1];
  const float* bq = (const float*)d_in[2];
  const float* Wk = (const float*)d_in[3];
  const float* bk = (const float*)d_in[4];
  const float* Wv = (const float*)d_in[5];
  const float* bv = (const float*)d_in[6];
  const float* Wo = (const float*)d_in[7];
  const float* bo = (const float*)d_in[8];

  char* ws = (char*)d_ws;
  __bf16* xb  = (__bf16*)(ws + OFF_XB);
  __bf16* wqb = (__bf16*)(ws + OFF_WQB);
  __bf16* wkb = (__bf16*)(ws + OFF_WKB);
  __bf16* wvb = (__bf16*)(ws + OFF_WVB);
  __bf16* wob = (__bf16*)(ws + OFF_WOB);
  __bf16* Qb  = (__bf16*)(ws + OFF_QB);
  __bf16* Kb  = (__bf16*)(ws + OFF_KB);
  __bf16* Vtb = (__bf16*)(ws + OFF_VTB);
  __bf16* ctxb= (__bf16*)(ws + OFF_CTX);

  const int nX = B_ * S_ * H_;          // 4,194,304
  const int nW = K_ * H_;               // 1,048,576 (each weight)
  k_convert<<<(nX / 4 + 255) / 256, 256, 0, stream>>>(x,  xb,  nX / 4);
  k_convert<<<(nW / 4 + 255) / 256, 256, 0, stream>>>(Wq, wqb, nW / 4);
  k_convert<<<(nW / 4 + 255) / 256, 256, 0, stream>>>(Wk, wkb, nW / 4);
  k_convert<<<(nW / 4 + 255) / 256, 256, 0, stream>>>(Wv, wvb, nW / 4);
  k_convert<<<(nW / 4 + 255) / 256, 256, 0, stream>>>(Wo, wob, nW / 4);

  dim3 gGemm(H_ / 128, MTOT_ / 128);    // 8 x 32
  k_gemm<1><<<gGemm, 256, 0, stream>>>(xb, wqb, bq, (void*)Qb);
  k_gemm<2><<<gGemm, 256, 0, stream>>>(xb, wkb, bk, (void*)Kb);
  k_gemm<3><<<gGemm, 256, 0, stream>>>(xb, wvb, bv, (void*)Vtb);

  dim3 gAttn(S_ / 64, B_ * G_ * HPG_);  // 32 x 32
  k_attn<<<gAttn, 128, 0, stream>>>(Qb, Kb, Vtb, ctxb);

  k_gemm<0><<<gGemm, 256, 0, stream>>>(ctxb, wob, bo, d_out);
}
